// SelfAttnGNN_36283883716929
// MI455X (gfx1250) — matrix operand order repaired and checked by reading, not compiled, not run on hardware
//
#include <hip/hip_runtime.h>
#include <hip/hip_bf16.h>

#define B_     1024
#define S_     16
#define H_     256
#define IN_    150
#define INP_   160          // K padded to multiple of 32
#define OUT_   22
#define HEADS_ 8
#define DH_    32
#define NN_    100000
#define G0_    (B_*(S_+1))       // 17408 groups in layer 0
#define M0_    (G0_*(S_+1))      // 295936 rows of h0

typedef __attribute__((ext_vector_type(16))) _Float16 v16h;
typedef __attribute__((ext_vector_type(8)))  _Float16 v8h;
typedef __attribute__((ext_vector_type(8)))  float    v8f;

// ---- CDNA5 async global->LDS copy (ASYNCcnt path, ISA 15.18.3 op 98) ----
__device__ __forceinline__ void async_g2l_b128(void* lds_dst, const void* gsrc) {
  const unsigned lds_off = (unsigned)(unsigned long long)(size_t)lds_dst; // low 32b = LDS offset
  asm volatile("global_load_async_to_lds_b128 %0, %1, off"
               :: "v"(lds_off), "v"((unsigned long long)(size_t)gsrc)
               : "memory");
}
__device__ __forceinline__ void wait_async0() {
  asm volatile("s_wait_asynccnt 0x0" ::: "memory");
}

// ---- WMMA fragment loaders (contiguous k = 16*(l/16) + j in both A and B, wave32) ----
// A: 16x32 (MxK) f16, row-major source. lane l: row = l%16,
// halves j=0..15 -> K = 16*(l/16) + j
__device__ __forceinline__ v16h load_a_frag(const _Float16* base, int ld) {
  const int lane = threadIdx.x & 31;
  const int row  = lane & 15;
  const int g    = lane >> 4;
  const v8h lo = *(const v8h*)(base + row * ld + 16 * g);
  const v8h hi = *(const v8h*)(base + row * ld + 16 * g + 8);
  return __builtin_shufflevector(lo, hi, 0,1,2,3,4,5,6,7,8,9,10,11,12,13,14,15);
}
// B: 32x16 (KxN) f16 from an N-major (transposed) weight: wT[n*ldK + k].
// lane l: col = l%16, halves j: K = 16*(l/16) + j -> 16 contiguous halves.
__device__ __forceinline__ v16h load_bT_frag(const _Float16* base, int ldK) {
  const int lane = threadIdx.x & 31;
  const int col  = lane & 15;
  const int g    = lane >> 4;
  const v8h* p = (const v8h*)(base + col * ldK + g * 16);
  const v8h lo = p[0];
  const v8h hi = p[1];
  return __builtin_shufflevector(lo, hi, 0,1,2,3,4,5,6,7,8,9,10,11,12,13,14,15);
}

// ---------------- conversion kernels ----------------
__global__ void cvt_pad_cols_kernel(_Float16* dst, const float* src, int rows, int sc, int dc) {
  const int i = blockIdx.x * 256 + threadIdx.x;
  if (i < rows * dc) {
    const int r = i / dc, c = i % dc;
    dst[i] = (c < sc) ? (_Float16)src[r * sc + c] : (_Float16)0.f;
  }
}
// dst: nmat matrices, each N x Kdst (N-major);  src: nmat matrices, each Ksrc x N
__global__ void cvt_transpose_kernel(_Float16* dst, const float* src,
                                     int Ksrc, int Kdst, int N, int nmat) {
  const int i = blockIdx.x * 256 + threadIdx.x;
  const int per = N * Kdst;
  if (i >= nmat * per) return;
  const int m = i / per, rem = i % per;
  const int n = rem / Kdst, k = rem % Kdst;
  dst[i] = (k < Ksrc) ? (_Float16)src[(size_t)m * Ksrc * N + (size_t)k * N + n]
                      : (_Float16)0.f;
}

// ---------------- index flattening ----------------
__global__ void ids_kernel(const int* __restrict__ batch,
                           const int* __restrict__ nbr2,   // B x S
                           const int* __restrict__ nbr1,   // (B*(S+1)) x S
                           int* __restrict__ ids0) {       // M0
  const int idx = blockIdx.x * 256 + threadIdx.x;
  if (idx >= M0_) return;
  const int r = idx / (S_ + 1);
  const int j = idx % (S_ + 1);
  int id;
  if (j == 0) {
    const int b = r / (S_ + 1), t = r % (S_ + 1);
    id = (t == 0) ? batch[b] : nbr2[b * S_ + (t - 1)];
  } else {
    id = nbr1[r * S_ + (j - 1)];
  }
  ids0[idx] = id;
}

// ---------------- gather + input GEMM + ReLU (64-row macro tile) ----------------
__global__ __launch_bounds__(256)
void gemm_in_kernel(const _Float16* __restrict__ nf16,   // NN x 160
                    const int* __restrict__ ids0,
                    const _Float16* __restrict__ wT,     // 256 x 160 (N-major)
                    const float* __restrict__ bias,      // 256
                    _Float16* __restrict__ out) {        // M0 x 256
  __shared__ __align__(16) _Float16 a_s[64 * INP_];      // 20 KB
  const int tile = blockIdx.x;                            // 64-row tile
  const int tid  = threadIdx.x;
  // stage 64 gathered rows via async global->LDS (20 x 16B chunks per row)
  for (int c = tid; c < 64 * (INP_ / 8); c += 256) {
    const int r = c / (INP_ / 8), cc = c % (INP_ / 8);
    const int id = ids0[tile * 64 + r];
    async_g2l_b128(a_s + (size_t)r * INP_ + cc * 8,
                   nf16 + (size_t)id * INP_ + cc * 8);
  }
  wait_async0();
  __syncthreads();
  const int wave = tid >> 5, lane = tid & 31;
  v8f acc[4][2] = {};
#pragma unroll
  for (int kc = 0; kc < INP_ / 32; ++kc) {
    const v16h b0 = load_bT_frag(wT + (wave * 2 + 0) * 16 * INP_ + kc * 32, INP_);
    const v16h b1 = load_bT_frag(wT + (wave * 2 + 1) * 16 * INP_ + kc * 32, INP_);
#pragma unroll
    for (int rt = 0; rt < 4; ++rt) {
      const v16h a = load_a_frag(a_s + rt * 16 * INP_ + kc * 32, INP_);
      acc[rt][0] = __builtin_amdgcn_wmma_f32_16x16x32_f16(false, a, false, b0, (short)0, acc[rt][0], false, false);
      acc[rt][1] = __builtin_amdgcn_wmma_f32_16x16x32_f16(false, a, false, b1, (short)0, acc[rt][1], false, false);
    }
  }
#pragma unroll
  for (int rt = 0; rt < 4; ++rt) {
#pragma unroll
    for (int i = 0; i < 8; ++i) {
      const int row = tile * 64 + rt * 16 + i + 8 * (lane >> 4);
      const int c0  = (wave * 2 + 0) * 16 + (lane & 15);
      const int c1  = (wave * 2 + 1) * 16 + (lane & 15);
      float v0 = acc[rt][0][i] + bias[c0]; v0 = v0 > 0.f ? v0 : 0.f;
      float v1 = acc[rt][1][i] + bias[c1]; v1 = v1 > 0.f ? v1 : 0.f;
      out[(size_t)row * H_ + c0] = (_Float16)v0;
      out[(size_t)row * H_ + c1] = (_Float16)v1;
    }
  }
}

// ---------------- fused QKV + attention (one workgroup per group) ----------------
__global__ __launch_bounds__(256)
void qkv_attn_kernel(const _Float16* __restrict__ x,   // groups*17*256
                     const _Float16* __restrict__ wqT, // 256x256 N-major
                     const _Float16* __restrict__ wkT,
                     const _Float16* __restrict__ wvT,
                     const float* __restrict__ bq,
                     const float* __restrict__ bk,
                     const float* __restrict__ bv,
                     _Float16* __restrict__ o_out) {   // groups*256
  __shared__ __align__(16) _Float16 x_s[17 * H_];
  __shared__ float kp_s[16 * H_];
  __shared__ float vp_s[16 * H_];
  __shared__ float qp_s[H_];
  __shared__ float att_s[HEADS_ * S_];
  const int g   = blockIdx.x;
  const int tid = threadIdx.x;
  const _Float16* xg = x + (size_t)g * 17 * H_;
  for (int i = tid; i < 17 * H_ / 8; i += 256)
    async_g2l_b128(x_s + i * 8, xg + i * 8);
  wait_async0();
  __syncthreads();

  const int wave = tid >> 5, lane = tid & 31;
  // waves 0-3 -> K projection, waves 4-7 -> V projection; 4 N-tiles each
  const _Float16* wmat = (wave < 4) ? wkT : wvT;
  const float*    bvec = (wave < 4) ? bk : bv;
  float*          dst  = (wave < 4) ? kp_s : vp_s;
  const int nbase = (wave & 3) * 4;
  v8f acc[4] = {};
#pragma unroll
  for (int kc = 0; kc < 8; ++kc) {
    const v16h a = load_a_frag(x_s + H_ + kc * 32, H_);   // kv = rows 1..16
#pragma unroll
    for (int nt = 0; nt < 4; ++nt) {
      const v16h b = load_bT_frag(wmat + (nbase + nt) * 16 * H_ + kc * 32, H_);
      acc[nt] = __builtin_amdgcn_wmma_f32_16x16x32_f16(false, a, false, b, (short)0, acc[nt], false, false);
    }
  }
#pragma unroll
  for (int nt = 0; nt < 4; ++nt) {
#pragma unroll
    for (int i = 0; i < 8; ++i) {
      const int row = i + 8 * (lane >> 4);
      const int col = (nbase + nt) * 16 + (lane & 15);
      dst[row * H_ + col] = acc[nt][i] + bvec[col];
    }
  }
  // q projection (M=1): thread tid computes column tid; wqT row tid is contiguous
  {
    float s = bq[tid];
    const v8h* wrow = (const v8h*)(wqT + (size_t)tid * H_);
    const v8h* xrow = (const v8h*)x_s;   // row 0 of x
#pragma unroll 4
    for (int k8 = 0; k8 < H_ / 8; ++k8) {
      const v8h w = wrow[k8];
      const v8h xv = xrow[k8];
#pragma unroll
      for (int j = 0; j < 8; ++j) s += (float)xv[j] * (float)w[j];
    }
    qp_s[tid] = s;
  }
  __syncthreads();
  // scores: 8 heads x 16 keys
  if (tid < HEADS_ * S_) {
    const int h = tid >> 4, k = tid & 15;
    float s = 0.f;
    for (int d = 0; d < DH_; ++d)
      s += qp_s[h * DH_ + d] * kp_s[k * H_ + h * DH_ + d];
    att_s[tid] = s * 0.17677669529663687f;   // 1/sqrt(32)
  }
  __syncthreads();
  if (tid < HEADS_) {
    float mx = -1e30f;
    for (int k = 0; k < S_; ++k) mx = fmaxf(mx, att_s[tid * S_ + k]);
    float sum = 0.f;
    for (int k = 0; k < S_; ++k) { float e = __expf(att_s[tid * S_ + k] - mx); att_s[tid * S_ + k] = e; sum += e; }
    const float inv = 1.f / sum;
    for (int k = 0; k < S_; ++k) att_s[tid * S_ + k] *= inv;
  }
  __syncthreads();
  {
    const int h = tid >> 5;   // DH_ == 32
    float o = 0.f;
#pragma unroll
    for (int k = 0; k < S_; ++k)
      o += att_s[h * S_ + k] * vp_s[k * H_ + tid];
    o_out[(size_t)g * H_ + tid] = (_Float16)o;
  }
}

// ------- batched GEMM + bias + residual + LayerNorm (32-row macro tile) -------
__global__ __launch_bounds__(256)
void gemm_res_ln_kernel(const _Float16* __restrict__ a,    // M x 256
                        const _Float16* __restrict__ wT,   // 256x256 N-major
                        const float* __restrict__ bias,
                        const _Float16* __restrict__ res,  // residual base
                        const int res_stride,              // per-row stride (elems)
                        const float* __restrict__ lng,
                        const float* __restrict__ lnb,
                        _Float16* __restrict__ out) {      // M x 256
  __shared__ __align__(16) _Float16 a_s[32 * H_];          // 16 KB
  __shared__ float r_s[32 * H_];                           // 32 KB
  __shared__ float mu_s[32], rs_s[32];
  const int tile = blockIdx.x;                             // 32-row tile
  const int tid  = threadIdx.x;
  for (int i = tid; i < 32 * H_ / 8; i += 256)
    async_g2l_b128(a_s + i * 8, a + (size_t)tile * 32 * H_ + i * 8);
  wait_async0();
  __syncthreads();
  const int wave = tid >> 5, lane = tid & 31;
  v8f acc[2][2] = {};
#pragma unroll
  for (int kc = 0; kc < 8; ++kc) {
    const v16h b0 = load_bT_frag(wT + (wave * 2 + 0) * 16 * H_ + kc * 32, H_);
    const v16h b1 = load_bT_frag(wT + (wave * 2 + 1) * 16 * H_ + kc * 32, H_);
#pragma unroll
    for (int rt = 0; rt < 2; ++rt) {
      const v16h af = load_a_frag(a_s + rt * 16 * H_ + kc * 32, H_);
      acc[rt][0] = __builtin_amdgcn_wmma_f32_16x16x32_f16(false, af, false, b0, (short)0, acc[rt][0], false, false);
      acc[rt][1] = __builtin_amdgcn_wmma_f32_16x16x32_f16(false, af, false, b1, (short)0, acc[rt][1], false, false);
    }
  }
#pragma unroll
  for (int rt = 0; rt < 2; ++rt) {
#pragma unroll
    for (int i = 0; i < 8; ++i) {
      const int row = rt * 16 + i + 8 * (lane >> 4);
      const int m   = tile * 32 + row;
      const int c0  = (wave * 2 + 0) * 16 + (lane & 15);
      const int c1  = (wave * 2 + 1) * 16 + (lane & 15);
      r_s[row * H_ + c0] = acc[rt][0][i] + bias[c0] + (float)res[(size_t)m * res_stride + c0];
      r_s[row * H_ + c1] = acc[rt][1][i] + bias[c1] + (float)res[(size_t)m * res_stride + c1];
    }
  }
  __syncthreads();
  if (tid < 32) {
    float s = 0.f;
    for (int c = 0; c < H_; ++c) s += r_s[tid * H_ + c];
    const float mu = s * (1.f / H_);
    float v = 0.f;
    for (int c = 0; c < H_; ++c) { const float d = r_s[tid * H_ + c] - mu; v += d * d; }
    mu_s[tid] = mu;
    rs_s[tid] = rsqrtf(v * (1.f / H_) + 1e-5f);
  }
  __syncthreads();
  for (int i = tid; i < 32 * H_; i += 256) {
    const int row = i >> 8, col = i & 255;
    const float val = (r_s[i] - mu_s[row]) * rs_s[row] * lng[col] + lnb[col];
    out[(size_t)(tile * 32 + row) * H_ + col] = (_Float16)val;
  }
}

// ---------------- output head + softmax ----------------
__global__ __launch_bounds__(256)
void out_kernel(const _Float16* __restrict__ h2,     // 1024 x 256
                const _Float16* __restrict__ woutT,  // 22 x 256 (N-major)
                const float* __restrict__ bout,
                float* __restrict__ out) {           // 1024 x 22
  const int row = blockIdx.x * 256 + threadIdx.x;
  if (row >= B_) return;
  const v8h* xrow = (const v8h*)(h2 + (size_t)row * H_);
  float lg[OUT_];
#pragma unroll
  for (int j = 0; j < OUT_; ++j) {
    float s = bout[j];
    const v8h* wrow = (const v8h*)(woutT + (size_t)j * H_);
    for (int k8 = 0; k8 < H_ / 8; ++k8) {
      const v8h w = wrow[k8];
      const v8h xv = xrow[k8];
#pragma unroll
      for (int t = 0; t < 8; ++t) s += (float)xv[t] * (float)w[t];
    }
    lg[j] = s;
  }
  float mx = lg[0];
#pragma unroll
  for (int j = 1; j < OUT_; ++j) mx = fmaxf(mx, lg[j]);
  float sum = 0.f;
#pragma unroll
  for (int j = 0; j < OUT_; ++j) { lg[j] = __expf(lg[j] - mx); sum += lg[j]; }
  const float inv = 1.f / sum;
#pragma unroll
  for (int j = 0; j < OUT_; ++j) out[(size_t)row * OUT_ + j] = lg[j] * inv;
}

// ---------------- host orchestration ----------------
extern "C" void kernel_launch(void* const* d_in, const int* in_sizes, int n_in,
                              void* d_out, int out_size, void* d_ws, size_t ws_size,
                              hipStream_t stream) {
  (void)in_sizes; (void)n_in; (void)out_size; (void)ws_size;
  const float* node_feats = (const float*)d_in[0];
  const int*   batch      = (const int*)d_in[1];
  const int*   nbr2       = (const int*)d_in[2];
  const int*   nbr1       = (const int*)d_in[3];
  const float* W_in       = (const float*)d_in[4];
  const float* b_in       = (const float*)d_in[5];
  const float* Wqkv       = (const float*)d_in[6];
  const float* bqkv       = (const float*)d_in[7];
  const float* Wo         = (const float*)d_in[8];
  const float* bo         = (const float*)d_in[9];
  const float* Wf         = (const float*)d_in[10];
  const float* bf         = (const float*)d_in[11];
  const float* ln_g       = (const float*)d_in[12];
  const float* ln_b       = (const float*)d_in[13];
  const float* W_out      = (const float*)d_in[14];
  const float* b_out      = (const float*)d_in[15];
  float* out = (float*)d_out;

  char* ws = (char*)d_ws;
  size_t off = 0;
  auto take = [&](size_t bytes) -> void* {
    void* p = ws + off;
    off = (off + bytes + 255) & ~(size_t)255;
    return p;
  };
  _Float16* nf16    = (_Float16*)take((size_t)NN_ * INP_ * 2);
  _Float16* w_inT   = (_Float16*)take((size_t)H_ * INP_ * 2);        // 256 x 160
  _Float16* wqkvT   = (_Float16*)take((size_t)2 * 3 * H_ * H_ * 2);  // 6 x 256 x 256
  _Float16* woT     = (_Float16*)take((size_t)2 * H_ * H_ * 2);
  _Float16* wfT     = (_Float16*)take((size_t)2 * H_ * H_ * 2);
  _Float16* woutT   = (_Float16*)take((size_t)OUT_ * H_ * 2);        // 22 x 256
  int*      ids0    = (int*)take((size_t)M0_ * 4);
  _Float16* h0      = (_Float16*)take((size_t)M0_ * H_ * 2);
  _Float16* o0      = (_Float16*)take((size_t)G0_ * H_ * 2);
  _Float16* hln0    = (_Float16*)take((size_t)G0_ * H_ * 2);
  _Float16* h1      = (_Float16*)take((size_t)G0_ * H_ * 2);
  _Float16* o1      = (_Float16*)take((size_t)B_ * H_ * 2);
  _Float16* hln1    = (_Float16*)take((size_t)B_ * H_ * 2);
  _Float16* h2      = (_Float16*)take((size_t)B_ * H_ * 2);

  const int MM = H_ * H_;   // 65536 elems per 256x256 matrix

  // f16 conversions; weights transposed to N-major for contiguous B-fragments
  cvt_pad_cols_kernel<<<(NN_ * INP_ + 255) / 256, 256, 0, stream>>>(nf16, node_feats, NN_, IN_, INP_);
  cvt_transpose_kernel<<<(H_ * INP_ + 255) / 256, 256, 0, stream>>>(w_inT, W_in, IN_, INP_, H_, 1);
  cvt_transpose_kernel<<<(6 * MM + 255) / 256, 256, 0, stream>>>(wqkvT, Wqkv, H_, H_, H_, 6);
  cvt_transpose_kernel<<<(2 * MM + 255) / 256, 256, 0, stream>>>(woT, Wo, H_, H_, H_, 2);
  cvt_transpose_kernel<<<(2 * MM + 255) / 256, 256, 0, stream>>>(wfT, Wf, H_, H_, H_, 2);
  cvt_transpose_kernel<<<(OUT_ * H_ + 255) / 256, 256, 0, stream>>>(woutT, W_out, H_, H_, OUT_, 1);

  // gather indices
  ids_kernel<<<(M0_ + 255) / 256, 256, 0, stream>>>(batch, nbr2, nbr1, ids0);

  // h0 = relu(gather @ W_in + b_in)
  gemm_in_kernel<<<M0_ / 64, 256, 0, stream>>>(nf16, ids0, w_inT, b_in, h0);

  // ---- layer 0 (17408 groups of 17 tokens from h0) ----
  qkv_attn_kernel<<<G0_, 256, 0, stream>>>(h0,
      wqkvT + 0 * MM, wqkvT + 1 * MM, wqkvT + 2 * MM,
      bqkv + 0 * H_, bqkv + 1 * H_, bqkv + 2 * H_, o0);
  gemm_res_ln_kernel<<<G0_ / 32, 256, 0, stream>>>(o0, woT + 0 * MM, bo + 0 * H_,
      h0, 17 * H_, ln_g + 0 * H_, ln_b + 0 * H_, hln0);
  gemm_res_ln_kernel<<<G0_ / 32, 256, 0, stream>>>(hln0, wfT + 0 * MM, bf + 0 * H_,
      hln0, H_, ln_g + 1 * H_, ln_b + 1 * H_, h1);

  // ---- layer 1 (1024 groups of 17 tokens from h1) ----
  qkv_attn_kernel<<<B_, 256, 0, stream>>>(h1,
      wqkvT + 3 * MM, wqkvT + 4 * MM, wqkvT + 5 * MM,
      bqkv + 3 * H_, bqkv + 4 * H_, bqkv + 5 * H_, o1);
  gemm_res_ln_kernel<<<B_ / 32, 256, 0, stream>>>(o1, woT + 1 * MM, bo + 1 * H_,
      h1, 17 * H_, ln_g + 2 * H_, ln_b + 2 * H_, hln1);
  gemm_res_ln_kernel<<<B_ / 32, 256, 0, stream>>>(hln1, wfT + 1 * MM, bf + 1 * H_,
      hln1, H_, ln_g + 3 * H_, ln_b + 3 * H_, h2);

  // output head + softmax
  out_kernel<<<(B_ + 255) / 256, 256, 0, stream>>>(h2, woutT, b_out, out);
}
